// VAGConditionalDecoder_46608985096434
// MI455X (gfx1250) — matrix operand order repaired and checked by reading, not compiled, not run on hardware
//
#include <hip/hip_runtime.h>
#include <hip/hip_bf16.h>
#include <stdint.h>

// ---------------------------------------------------------------------------
// VAG conditional decoder (attention GRU) for MI455X / gfx1250.
// bf16 WMMA 16x16x32 (f32 accum). Dominant GEMM (64x32000x512 per step) uses
// TDM tensor_load_to_lds double-buffered weight staging (64-K stages) +
// ds_load_b128 fragments. Whole bf16 working set (~60MB) is L2-resident.
// ---------------------------------------------------------------------------

typedef __attribute__((ext_vector_type(16))) __bf16   v16bf;
typedef __attribute__((ext_vector_type(8)))  float    v8f;
typedef __attribute__((ext_vector_type(8)))  uint32_t v8u;
typedef __attribute__((ext_vector_type(4)))  uint32_t v4u;
typedef __attribute__((ext_vector_type(8)))  int      v8i;
typedef __attribute__((ext_vector_type(4)))  int      v4i;

#define S_LEN   64
#define B_SZ    64
#define C_DIM   512
#define H_DIM   512
#define E_DIM   512
#define V_SZ    32000
#define T_STEPS 48
#define NEGV    (-100000000.0f)

// ------------------------------- helpers -----------------------------------

__device__ __forceinline__ uint16_t f2bf(float f) {
  uint32_t u = __float_as_uint(f);
  u += 0x7FFFu + ((u >> 16) & 1u);        // round-to-nearest-even
  return (uint16_t)(u >> 16);
}
__device__ __forceinline__ float bf2f(uint16_t h) {
  return __uint_as_float(((uint32_t)h) << 16);
}
__device__ __forceinline__ float sigm(float x) {
  return 1.0f / (1.0f + __expf(-x));
}

__device__ __forceinline__ v8f wmma_bf16(v16bf a, v16bf b, v8f c) {
  // (neg_a, A, neg_b, B, c_mod, C, reuse_a, reuse_b)
  return __builtin_amdgcn_wmma_f32_16x16x32_bf16(false, a, false, b,
                                                 (short)0, c, false, false);
}

// A-matrix fragment (16x32 bf16, M x K). contiguous-K order, same as B:
//  lane l: row m = l%16; half = l>>4
//  VGPR r: K = 2r + (half?16:0)
__device__ __forceinline__ v16bf load_a_frag(const uint16_t* A, int lda,
                                             int mbase, int kbase, int lane) {
  const int m = mbase + (lane & 15);
  const int half = lane >> 4;
  const uint16_t* row = A + (size_t)m * lda + kbase;
  v8u u;
#pragma unroll
  for (int r = 0; r < 8; ++r) {
    int k = 2 * r + (half ? 16 : 0);
    u[r] = *(const uint32_t*)(row + k);
  }
  return __builtin_bit_cast(v16bf, u);
}

// B-matrix fragment (32x16 bf16, K x N) from row-major W[N,K]:
//  lane l: col n = l%16; half = l>>4; VGPR r: K = 2r + (half?16:0)
__device__ __forceinline__ v16bf load_b_frag(const uint16_t* W, int ldw,
                                             int nbase, int kbase, int lane) {
  const int n = nbase + (lane & 15);
  const int half = lane >> 4;
  const uint16_t* row = W + (size_t)n * ldw + kbase + (half ? 16 : 0);
  v8u u;
#pragma unroll
  for (int r = 0; r < 8; ++r) u[r] = *(const uint32_t*)(row + 2 * r);
  return __builtin_bit_cast(v16bf, u);
}

// B fragment from an LDS tile packed as rows of 64 bf16 (128B per N-row).
// ksub selects the 32-K subchunk. Per lane: 32 contiguous bytes -> 2x ds_load_b128.
__device__ __forceinline__ v16bf load_b_frag_lds64(const uint16_t* buf,
                                                   int nloc, int ksub, int lane) {
  const int n = nloc + (lane & 15);
  const int half = lane >> 4;
  const uint16_t* row = buf + n * 64 + ksub + (half ? 16 : 0);
  v8u u;
#pragma unroll
  for (int r = 0; r < 8; ++r) u[r] = *(const uint32_t*)(row + 2 * r);
  return __builtin_bit_cast(v16bf, u);
}

// TDM 2D tile load: wk (K, contiguous) x 128 (N rows) bf16 tile -> packed LDS.
// D# per CDNA5 ISA ch.8: group0 {count, lds_addr, global_addr, type=2},
// group1 {data_size=2B, tensor dims, tile dims, dim0 stride}. 2D -> g2=g3=0.
// amdgpu-toolchain (clang-23) builtin: (v4u, v8i, v4i, v4i, v8i, i32 cpol).
__device__ __forceinline__ void tdm_load_tile(uint32_t lds_off,
                                              const uint16_t* gptr,
                                              uint32_t wk,
                                              uint32_t stride_elems) {
  uint64_t ga = (uint64_t)(uintptr_t)gptr;
  uint32_t ga_lo = __builtin_amdgcn_readfirstlane((uint32_t)ga);
  uint32_t ga_hi = __builtin_amdgcn_readfirstlane((uint32_t)(ga >> 32));
  lds_off = __builtin_amdgcn_readfirstlane(lds_off);
  v4u g0;
  g0[0] = 1u;                                   // count=1 (valid user D#)
  g0[1] = lds_off;                              // lds_addr (bytes)
  g0[2] = ga_lo;                                // global_addr[31:0]
  g0[3] = (ga_hi & 0x01ffffffu) | (2u << 30);   // global_addr[56:32] | type=2
  v8i g1;
  g1[0] = (int)(1u << 16);                      // data_size=1 -> 2 bytes
  g1[1] = (int)(wk << 16);                      // tensor_dim0 (lo16)
  g1[2] = (int)(128u << 16);                    // td0_hi=0 | tensor_dim1=128
  g1[3] = (int)(wk << 16);                      // td1_hi=0 | tile_dim0=wk
  g1[4] = (int)(128u);                          // tile_dim1=128, tile_dim2=0
  g1[5] = (int)stride_elems;                    // tensor_dim0_stride (lo32)
  g1[6] = 0;                                    // stride hi / dim1_stride lo
  g1[7] = 0;
  v4i z4 = {0, 0, 0, 0};
  v8i z8 = {0, 0, 0, 0, 0, 0, 0, 0};
  __builtin_amdgcn_tensor_load_to_lds(g0, g1, z4, z4, z8, 0);
}

// ------------------------------- kernels -----------------------------------

__global__ void k_zero(float* p, int n) {
  int i = blockIdx.x * blockDim.x + threadIdx.x;
  if (i < n) p[i] = 0.0f;
}

__global__ void k_cvt_bf16(const float* __restrict__ src,
                           uint16_t* __restrict__ dst, size_t n) {
  size_t i = (size_t)blockIdx.x * blockDim.x + threadIdx.x;
  size_t stride = (size_t)gridDim.x * blockDim.x;
  for (; i < n; i += stride) dst[i] = f2bf(src[i]);
}

__global__ void k_mean_mix(const float* __restrict__ ctx,
                           const float* __restrict__ mask,
                           const float* __restrict__ txt,
                           uint16_t* __restrict__ mixed) {
  int idx = blockIdx.x * blockDim.x + threadIdx.x;   // B*C
  if (idx >= B_SZ * C_DIM) return;
  int b = idx / C_DIM, c = idx % C_DIM;
  float cs = 0.f, ms = 0.f;
  for (int s = 0; s < S_LEN; ++s) {
    cs += ctx[((size_t)s * B_SZ + b) * C_DIM + c];
    ms += mask[s * B_SZ + b];
  }
  mixed[idx] = f2bf(0.5f * (cs / ms) + 0.5f * txt[idx]);
}

__global__ void k_embed(const float* __restrict__ emb,
                        const int* __restrict__ y,
                        uint16_t* __restrict__ xseq) {
  size_t idx = (size_t)blockIdx.x * blockDim.x + threadIdx.x;  // T*B*E
  int e = (int)(idx % E_DIM);
  int tb = (int)(idx / E_DIM);
  int tok = y[tb];
  xseq[idx] = f2bf(emb[(size_t)tok * E_DIM + e]);
}

// Generic GEMM: C[M,N] = act(A[M,K](bf16) @ W[N,K]^T(bf16) + bias)
// Block 256 thr/8 waves; tile 32(M)x256(N); wave: 1 Mtile x 4 Ntiles.
// Register double-buffered over K so loads overlap WMMA.
template <int ACT>
__global__ void gemm_bf16(const uint16_t* __restrict__ A,
                          const uint16_t* __restrict__ W,
                          const float* __restrict__ bias,
                          float* __restrict__ Cf, uint16_t* __restrict__ Cb,
                          int M, int N, int K) {
  const int lane = threadIdx.x & 31;
  const int wave = threadIdx.x >> 5;
  const int mbase = blockIdx.y * 32 + (wave & 1) * 16;
  const int nbase = blockIdx.x * 256 + (wave >> 1) * 64;

  v8f acc[4] = {};
  v16bf a = load_a_frag(A, K, mbase, 0, lane);
  v16bf b[4];
#pragma unroll
  for (int t = 0; t < 4; ++t) b[t] = load_b_frag(W, K, nbase + t * 16, 0, lane);

  for (int k = 32; k < K; k += 32) {
    __builtin_prefetch((const void*)(W + (size_t)(nbase + (lane & 15)) * K + k + 32), 0, 1);
    v16bf an = load_a_frag(A, K, mbase, k, lane);
    v16bf bn[4];
#pragma unroll
    for (int t = 0; t < 4; ++t) bn[t] = load_b_frag(W, K, nbase + t * 16, k, lane);
#pragma unroll
    for (int t = 0; t < 4; ++t) acc[t] = wmma_bf16(a, b[t], acc[t]);
    a = an;
#pragma unroll
    for (int t = 0; t < 4; ++t) b[t] = bn[t];
  }
#pragma unroll
  for (int t = 0; t < 4; ++t) acc[t] = wmma_bf16(a, b[t], acc[t]);

  const int nn = lane & 15, half = lane >> 4;
#pragma unroll
  for (int t = 0; t < 4; ++t) {
#pragma unroll
    for (int r = 0; r < 8; ++r) {
      int m = mbase + r + 8 * half;     // D layout: VGPR r -> M = r + 8*half
      int n = nbase + t * 16 + nn;
      float v = acc[t][r] + (bias ? bias[n] : 0.0f);
      if (ACT) v = tanhf(v);
      if (Cf) Cf[(size_t)m * N + n] = v;
      if (Cb) Cb[(size_t)m * N + n] = f2bf(v);
    }
  }
}

// Fused output projection + softmax denominator, TDM-staged weights:
//   z[64,V] = logit(bf16) @ W_o2p^T + b ; rowsum[b] += sum_n exp(z[b,n])
// Block 256 thr/8 waves covers 4 Mtiles x 128 cols (grid 250).
// Weight tiles (128 x 64 bf16 = 16KB) double-buffered in LDS via TDM;
// 8 WMMAs per barrier/tensor-wait stage.
__global__ void k_lse_gemm(const uint16_t* __restrict__ A,
                           const uint16_t* __restrict__ W,
                           const float* __restrict__ bias,
                           float* __restrict__ rowsum, int K) {
  __shared__ uint16_t wtile[2][128 * 64];   // 2 x 16KB
  const int lane = threadIdx.x & 31;
  const int wave = threadIdx.x >> 5;
  const int mbase = (wave & 3) * 16;
  const int nloc  = (wave >> 2) * 64;
  const int nst = K / 64;                   // 8 stages
  const uint16_t* Wblk = W + (size_t)(blockIdx.x * 128) * K;

  if (wave == 0)
    tdm_load_tile((uint32_t)(uintptr_t)&wtile[0][0], Wblk, 64u, (uint32_t)K);

  v8f acc[4] = {};
  for (int s = 0; s < nst; ++s) {
    if (wave == 0) __builtin_amdgcn_s_wait_tensorcnt(0);  // stage s resident
    __syncthreads();
    if (wave == 0 && s + 1 < nst)
      tdm_load_tile((uint32_t)(uintptr_t)&wtile[(s + 1) & 1][0],
                    Wblk + (s + 1) * 64, 64u, (uint32_t)K);

    // Issue all stage loads first (A from global, 8 B fragments from LDS),
    // then the 8 WMMAs, so waits don't interleave with the XDL stream.
    v16bf a0 = load_a_frag(A, K, mbase, s * 64, lane);
    v16bf a1 = load_a_frag(A, K, mbase, s * 64 + 32, lane);
    const uint16_t* wb = &wtile[s & 1][0];
    v16bf b0[4], b1[4];
#pragma unroll
    for (int t = 0; t < 4; ++t) b0[t] = load_b_frag_lds64(wb, nloc + t * 16, 0, lane);
#pragma unroll
    for (int t = 0; t < 4; ++t) b1[t] = load_b_frag_lds64(wb, nloc + t * 16, 32, lane);
#pragma unroll
    for (int t = 0; t < 4; ++t) acc[t] = wmma_bf16(a0, b0[t], acc[t]);
#pragma unroll
    for (int t = 0; t < 4; ++t) acc[t] = wmma_bf16(a1, b1[t], acc[t]);
    __syncthreads();                        // all reads done before reuse
  }

  const int nn = lane & 15, half = lane >> 4;
  const int nbase_g = blockIdx.x * 128 + nloc;
#pragma unroll
  for (int r = 0; r < 8; ++r) {
    float es = 0.f;
#pragma unroll
    for (int t = 0; t < 4; ++t) {
      int n = nbase_g + t * 16 + nn;
      es += __expf(acc[t][r] + bias[n]);    // logits ~O(5): exp safe in f32
    }
#pragma unroll
    for (int off = 1; off < 16; off <<= 1) es += __shfl_xor(es, off, 16);
    if (nn == 0) atomicAdd(&rowsum[mbase + r + 8 * half], es);
  }
}

__global__ void k_gru_combine(const float* __restrict__ gi,
                              const float* __restrict__ gh,
                              const float* __restrict__ hp,
                              float* __restrict__ hf,
                              uint16_t* __restrict__ hb) {
  int idx = blockIdx.x * blockDim.x + threadIdx.x;   // B*H
  int b = idx / H_DIM, j = idx % H_DIM;
  const float* gib = gi + (size_t)b * 3 * H_DIM;
  const float* ghb = gh + (size_t)b * 3 * H_DIM;
  float r = sigm(gib[j] + ghb[j]);
  float z = sigm(gib[H_DIM + j] + ghb[H_DIM + j]);
  float n = tanhf(gib[2 * H_DIM + j] + r * ghb[2 * H_DIM + j]);
  float h = (1.0f - z) * n + z * hp[idx];
  hf[idx] = h;
  hb[idx] = f2bf(h);
}

__global__ void k_attn_score(const float* __restrict__ ctx_proj,
                             const float* __restrict__ q,
                             const float* __restrict__ wm,
                             const float* __restrict__ mask,
                             float* __restrict__ scores) {
  __shared__ float red[128];
  const int sb = blockIdx.x;            // s*B + b
  const int b = sb % B_SZ;
  const int tid = threadIdx.x;
  const float* cp = ctx_proj + (size_t)sb * C_DIM;
  const float* qb = q + (size_t)b * C_DIM;
  float acc = 0.f;
  for (int c = tid; c < C_DIM; c += 128) acc += tanhf(cp[c] + qb[c]) * wm[c];
  red[tid] = acc;
  __syncthreads();
  for (int off = 64; off > 0; off >>= 1) {
    if (tid < off) red[tid] += red[tid + off];
    __syncthreads();
  }
  if (tid == 0) scores[sb] = (mask[sb] > 0.0f) ? red[0] : NEGV;
}

__global__ void k_softmax(const float* __restrict__ scores,
                          float* __restrict__ alpha) {
  __shared__ float sm[S_LEN];
  const int b = blockIdx.x, s = threadIdx.x;
  float v = scores[s * B_SZ + b];
  sm[s] = v;
  __syncthreads();
  for (int off = 32; off > 0; off >>= 1) {
    if (s < off) sm[s] = fmaxf(sm[s], sm[s + off]);
    __syncthreads();
  }
  float mx = sm[0];
  __syncthreads();
  float e = __expf(v - mx);
  sm[s] = e;
  __syncthreads();
  for (int off = 32; off > 0; off >>= 1) {
    if (s < off) sm[s] += sm[s + off];
    __syncthreads();
  }
  alpha[s * B_SZ + b] = e / sm[0];
}

__global__ void k_attn_ctx(const float* __restrict__ alpha,
                           const float* __restrict__ ctx,
                           uint16_t* __restrict__ z1) {
  int idx = blockIdx.x * blockDim.x + threadIdx.x;   // B*C
  int b = idx / C_DIM, c = idx % C_DIM;
  float acc = 0.f;
  for (int s = 0; s < S_LEN; ++s)
    acc += alpha[s * B_SZ + b] * ctx[((size_t)s * B_SZ + b) * C_DIM + c];
  z1[idx] = f2bf(acc);
}

__global__ void k_tgt(const uint16_t* __restrict__ logit,
                      const uint16_t* __restrict__ W,
                      const float* __restrict__ bias,
                      const int* __restrict__ y, int t,
                      float* __restrict__ ztgt) {
  __shared__ float red[256];
  const int b = blockIdx.x, tid = threadIdx.x;
  const int tok = y[(t + 1) * B_SZ + b];
  const uint16_t* lb = logit + (size_t)b * E_DIM;
  const uint16_t* wr = W + (size_t)tok * E_DIM;
  float acc = 0.f;
  for (int e = tid; e < E_DIM; e += 256) acc += bf2f(lb[e]) * bf2f(wr[e]);
  red[tid] = acc;
  __syncthreads();
  for (int off = 128; off > 0; off >>= 1) {
    if (tid < off) red[tid] += red[tid + off];
    __syncthreads();
  }
  if (tid == 0) ztgt[b] = red[0] + bias[tok];
}

__global__ void k_loss(const float* __restrict__ rowsum,
                       const float* __restrict__ ztgt,
                       const int* __restrict__ y, int t,
                       float* __restrict__ out) {
  __shared__ float red[B_SZ];
  const int b = threadIdx.x;
  const int tok = y[(t + 1) * B_SZ + b];
  float lp = ztgt[b] - __logf(rowsum[b]);
  red[b] = (tok != 0) ? -lp : 0.0f;
  __syncthreads();
  for (int off = 32; off > 0; off >>= 1) {
    if (b < off) red[b] += red[b + off];
    __syncthreads();
  }
  if (b == 0) out[0] += red[0];   // stream-serialized across steps
}

// ------------------------------- launch ------------------------------------

extern "C" void kernel_launch(void* const* d_in, const int* in_sizes, int n_in,
                              void* d_out, int out_size, void* d_ws, size_t ws_size,
                              hipStream_t stream) {
  const float* ctx      = (const float*)d_in[0];
  const float* ctx_mask = (const float*)d_in[1];
  const float* txt_ctx  = (const float*)d_in[2];
  const int*   y        = (const int*)d_in[3];
  const float* emb      = (const float*)d_in[4];
  const float* W_c2c    = (const float*)d_in[5];
  const float* W_h2c    = (const float*)d_in[6];
  const float* w_mlp    = (const float*)d_in[7];
  const float* W_c2h    = (const float*)d_in[8];
  const float* W_dec    = (const float*)d_in[9];
  const float* W_ih0    = (const float*)d_in[10];
  const float* W_hh0    = (const float*)d_in[11];
  const float* b_ih0    = (const float*)d_in[12];
  const float* b_hh0    = (const float*)d_in[13];
  const float* W_ih1    = (const float*)d_in[14];
  const float* W_hh1    = (const float*)d_in[15];
  const float* b_ih1    = (const float*)d_in[16];
  const float* b_hh1    = (const float*)d_in[17];
  const float* W_h2o    = (const float*)d_in[18];
  const float* b_h2o    = (const float*)d_in[19];
  const float* W_o2p    = (const float*)d_in[20];
  const float* b_o2p    = (const float*)d_in[21];
  float* out = (float*)d_out;

  char* p = (char*)d_ws;
  auto alloc = [&](size_t bytes) -> void* {
    void* r = (void*)p;
    p += (bytes + 255) & ~(size_t)255;
    return r;
  };
  uint16_t* wb_c2c  = (uint16_t*)alloc((size_t)C_DIM * C_DIM * 2);
  uint16_t* wb_h2c  = (uint16_t*)alloc((size_t)C_DIM * H_DIM * 2);
  uint16_t* wb_c2h  = (uint16_t*)alloc((size_t)H_DIM * C_DIM * 2);
  uint16_t* wb_dec  = (uint16_t*)alloc((size_t)H_DIM * C_DIM * 2);
  uint16_t* wb_h2o  = (uint16_t*)alloc((size_t)E_DIM * H_DIM * 2);
  uint16_t* wb_ih0  = (uint16_t*)alloc((size_t)3 * H_DIM * E_DIM * 2);
  uint16_t* wb_hh0  = (uint16_t*)alloc((size_t)3 * H_DIM * H_DIM * 2);
  uint16_t* wb_ih1  = (uint16_t*)alloc((size_t)3 * H_DIM * H_DIM * 2);
  uint16_t* wb_hh1  = (uint16_t*)alloc((size_t)3 * H_DIM * H_DIM * 2);
  uint16_t* wb_o2p  = (uint16_t*)alloc((size_t)V_SZ * E_DIM * 2);
  uint16_t* ctx_bf  = (uint16_t*)alloc((size_t)S_LEN * B_SZ * C_DIM * 2);
  uint16_t* xseq_bf = (uint16_t*)alloc((size_t)T_STEPS * B_SZ * E_DIM * 2);
  uint16_t* mixed_bf= (uint16_t*)alloc((size_t)B_SZ * C_DIM * 2);
  uint16_t* h_bf    = (uint16_t*)alloc((size_t)B_SZ * H_DIM * 2);
  uint16_t* h1_bf   = (uint16_t*)alloc((size_t)B_SZ * H_DIM * 2);
  uint16_t* z1_bf   = (uint16_t*)alloc((size_t)B_SZ * C_DIM * 2);
  uint16_t* zt_bf   = (uint16_t*)alloc((size_t)B_SZ * H_DIM * 2);
  uint16_t* logit_bf= (uint16_t*)alloc((size_t)B_SZ * E_DIM * 2);
  float* ctx_proj = (float*)alloc((size_t)S_LEN * B_SZ * C_DIM * 4);
  float* gi       = (float*)alloc((size_t)B_SZ * 3 * H_DIM * 4);
  float* gh       = (float*)alloc((size_t)B_SZ * 3 * H_DIM * 4);
  float* qbuf     = (float*)alloc((size_t)B_SZ * C_DIM * 4);
  float* scores   = (float*)alloc((size_t)S_LEN * B_SZ * 4);
  float* alpha    = (float*)alloc((size_t)S_LEN * B_SZ * 4);
  float* h_f      = (float*)alloc((size_t)B_SZ * H_DIM * 4);
  float* h1_f     = (float*)alloc((size_t)B_SZ * H_DIM * 4);
  float* rowsum   = (float*)alloc((size_t)B_SZ * 4);
  float* ztgt     = (float*)alloc((size_t)B_SZ * 4);

  auto cvt = [&](const float* s, uint16_t* d, size_t n) {
    k_cvt_bf16<<<1024, 256, 0, stream>>>(s, d, n);
  };
  auto gemm = [&](const uint16_t* A, const uint16_t* W, const float* bias,
                  float* Cf, uint16_t* Cb, int M, int N, int K, int act) {
    dim3 g(N / 256, M / 32);
    if (act) gemm_bf16<1><<<g, 256, 0, stream>>>(A, W, bias, Cf, Cb, M, N, K);
    else     gemm_bf16<0><<<g, 256, 0, stream>>>(A, W, bias, Cf, Cb, M, N, K);
  };

  // ---- setup ----
  k_zero<<<1, 1, 0, stream>>>(out, 1);
  cvt(W_c2c, wb_c2c, (size_t)C_DIM * C_DIM);
  cvt(W_h2c, wb_h2c, (size_t)C_DIM * H_DIM);
  cvt(W_c2h, wb_c2h, (size_t)H_DIM * C_DIM);
  cvt(W_dec, wb_dec, (size_t)H_DIM * C_DIM);
  cvt(W_h2o, wb_h2o, (size_t)E_DIM * H_DIM);
  cvt(W_ih0, wb_ih0, (size_t)3 * H_DIM * E_DIM);
  cvt(W_hh0, wb_hh0, (size_t)3 * H_DIM * H_DIM);
  cvt(W_ih1, wb_ih1, (size_t)3 * H_DIM * H_DIM);
  cvt(W_hh1, wb_hh1, (size_t)3 * H_DIM * H_DIM);
  cvt(W_o2p, wb_o2p, (size_t)V_SZ * E_DIM);
  cvt(ctx,   ctx_bf, (size_t)S_LEN * B_SZ * C_DIM);

  k_mean_mix<<<(B_SZ * C_DIM) / 256, 256, 0, stream>>>(ctx, ctx_mask, txt_ctx, mixed_bf);
  gemm(mixed_bf, wb_dec, nullptr, h_f, h_bf, B_SZ, H_DIM, C_DIM, 1);   // h0
  k_embed<<<(T_STEPS * B_SZ * E_DIM) / 256, 256, 0, stream>>>(emb, y, xseq_bf);
  gemm(ctx_bf, wb_c2c, nullptr, ctx_proj, nullptr, S_LEN * B_SZ, C_DIM, C_DIM, 0);

  // ---- recurrence ----
  for (int t = 0; t < T_STEPS; ++t) {
    const uint16_t* xt = xseq_bf + (size_t)t * B_SZ * E_DIM;
    // GRU layer 0
    gemm(xt,   wb_ih0, b_ih0, gi, nullptr, B_SZ, 3 * H_DIM, E_DIM, 0);
    gemm(h_bf, wb_hh0, b_hh0, gh, nullptr, B_SZ, 3 * H_DIM, H_DIM, 0);
    k_gru_combine<<<(B_SZ * H_DIM) / 256, 256, 0, stream>>>(gi, gh, h_f, h1_f, h1_bf);
    // attention
    gemm(h1_bf, wb_h2c, nullptr, qbuf, nullptr, B_SZ, C_DIM, H_DIM, 0);
    k_attn_score<<<S_LEN * B_SZ, 128, 0, stream>>>(ctx_proj, qbuf, w_mlp, ctx_mask, scores);
    k_softmax<<<B_SZ, S_LEN, 0, stream>>>(scores, alpha);
    k_attn_ctx<<<(B_SZ * C_DIM) / 256, 256, 0, stream>>>(alpha, ctx, z1_bf);
    gemm(z1_bf, wb_c2h, nullptr, nullptr, zt_bf, B_SZ, H_DIM, C_DIM, 0);
    // GRU layer 1
    gemm(zt_bf, wb_ih1, b_ih1, gi, nullptr, B_SZ, 3 * H_DIM, H_DIM, 0);
    gemm(h1_bf, wb_hh1, b_hh1, gh, nullptr, B_SZ, 3 * H_DIM, H_DIM, 0);
    k_gru_combine<<<(B_SZ * H_DIM) / 256, 256, 0, stream>>>(gi, gh, h1_f, h_f, h_bf);
    // output head
    gemm(h_bf, wb_h2o, b_h2o, nullptr, logit_bf, B_SZ, E_DIM, H_DIM, 1);
    k_zero<<<1, 64, 0, stream>>>(rowsum, B_SZ);
    k_lse_gemm<<<V_SZ / 128, 256, 0, stream>>>(logit_bf, wb_o2p, b_o2p, rowsum, E_DIM);
    k_tgt<<<B_SZ, 256, 0, stream>>>(logit_bf, wb_o2p, b_o2p, y, t, ztgt);
    k_loss<<<1, B_SZ, 0, stream>>>(rowsum, ztgt, y, t, out);
  }
}
